// OKTNet_56075093016912
// MI455X (gfx1250) — matrix operand order repaired and checked by reading, not compiled, not run on hardware
//
#include <hip/hip_runtime.h>
#include <hip/hip_bf16.h>
#include <stdint.h>
#include <stddef.h>

// ---------------- model dims ----------------
#define B_   512
#define S_   500
#define DH_  128
#define HID_ 256

typedef __bf16 bf16;
typedef __attribute__((ext_vector_type(16))) __bf16 bf16x16;
typedef __attribute__((ext_vector_type(8)))  float  f32x8;

// f32 -> bf16 round-to-nearest-even (bit manipulation, no __bf16 arithmetic)
__device__ __forceinline__ bf16 f2bf(float f) {
  union { float f; uint32_t u; } a; a.f = f;
  uint32_t r = a.u + 0x7FFFu + ((a.u >> 16) & 1u);
  uint16_t h = (uint16_t)(r >> 16);
  bf16 b; __builtin_memcpy(&b, &h, 2);
  return b;
}
__device__ __forceinline__ float sigm(float x) { return 1.f / (1.f + __expf(-x)); }

// ---------------------------------------------------------------------------
// A-operand swizzle for V_WMMA_F32_16X16X32_BF16 (16-bit A 16x32, ISA 7.12.2):
//   lane = (m&15) + 16*half, half = (k>>3)&1
//   elem e = 2*v + (k&1),  v = (k>=16 ? 4 : 0) + ((k&7)>>1)
// Even k and k+1 map to consecutive elements of the same lane -> 4B stores.
// Tile (mt,kt) of a [64 x 32*ktStride] A lives at ((mt*ktStride)+kt)*512 elems.
// ---------------------------------------------------------------------------
__device__ __forceinline__ int a_swz(int m, int k, int ktStride) {
  int mt = m >> 4, mr = m & 15;
  int kt = k >> 5, kk = k & 31;
  int half = (kk >> 3) & 1;
  int kb = kk & 7;
  int v = ((kk >= 16) ? 4 : 0) + (kb >> 1);
  int e = 2 * v + (kb & 1);
  return ((mt * ktStride + kt) << 9) + ((mr + (half << 4)) << 4) + e;
}
__device__ __forceinline__ void a_store2(bf16* Ab, int ktStride, int m, int k,
                                         float x0, float x1) {
  union { uint32_t u; bf16 b[2]; } p;
  p.b[0] = f2bf(x0); p.b[1] = f2bf(x1);
  *(uint32_t*)(Ab + a_swz(m, k, ktStride)) = p.u;
}
__device__ __forceinline__ void a_store2bf(bf16* Ab, int ktStride, int m, int k,
                                           uint32_t two) {
  *(uint32_t*)(Ab + a_swz(m, k, ktStride)) = two;
}

// ---------------------------------------------------------------------------
// Pack weight [K,N] (row-major f32, reference stores [in,out]) into bf16
// B-operand tiles: tile (kt,nt) at ((kt*(N/16))+nt)*512, per-lane 16 contig
// elems; lanes 0-15: K 0..7,16..23, lanes 16-31: K 8..15,24..31 (A-swizzle order).
// ---------------------------------------------------------------------------
__global__ __launch_bounds__(256)
void pack_b(const float* __restrict__ src, bf16* __restrict__ dst, int K, int N) {
  int d = blockIdx.x * 256 + threadIdx.x;
  if (d >= K * N) return;
  int tile = d >> 9, idx = d & 511;
  int lane = idx >> 4, e = idx & 15;
  int ntc = N >> 4;
  int nt = tile % ntc, kt = tile / ntc;
  int half = lane >> 4, v = e >> 1, sub = e & 1;
  int k = kt * 32 + ((e < 8) ? (8 * half + 2 * v + sub) : (16 + 8 * half + 2 * v + sub - 8));
  int n = nt * 16 + (lane & 15);
  dst[d] = f2bf(src[k * N + n]);
}

// ---------------------------------------------------------------------------
// Workgroup GEMM: C[64 x 16*ntCount] = A[64 x 32*ktiles] * B
// 256 threads = 8 waves; wave w owns N-tiles w, w+8, ...; 4 M-tiles each.
// A in LDS (swizzled), B streamed from global (packed), C to LDS f32.
// ---------------------------------------------------------------------------
__device__ __forceinline__ void gemm64(const bf16* Ab, int ktStride,
                                       const bf16* __restrict__ Bg,
                                       int ktiles, int ntCount,
                                       float* Cb, int cCols, int cOff) {
  const int wave = threadIdx.x >> 5, lane = threadIdx.x & 31;
  for (int nt = wave; nt < ntCount; nt += 8) {
    f32x8 c0 = {}, c1 = {}, c2 = {}, c3 = {};
    for (int kt = 0; kt < ktiles; ++kt) {
      if (kt + 1 < ktiles)
        __builtin_prefetch(Bg + (size_t)((kt + 1) * ntCount + nt) * 512 + lane * 16, 0, 1);
      const bf16x16 bt = *(const bf16x16*)(Bg + (size_t)(kt * ntCount + nt) * 512 + lane * 16);
      const bf16x16 a0 = *(const bf16x16*)(Ab + ((0 * ktStride + kt) << 9) + lane * 16);
      const bf16x16 a1 = *(const bf16x16*)(Ab + ((1 * ktStride + kt) << 9) + lane * 16);
      const bf16x16 a2 = *(const bf16x16*)(Ab + ((2 * ktStride + kt) << 9) + lane * 16);
      const bf16x16 a3 = *(const bf16x16*)(Ab + ((3 * ktStride + kt) << 9) + lane * 16);
      c0 = __builtin_amdgcn_wmma_f32_16x16x32_bf16(false, a0, false, bt, (short)0, c0, false, false);
      c1 = __builtin_amdgcn_wmma_f32_16x16x32_bf16(false, a1, false, bt, (short)0, c1, false, false);
      c2 = __builtin_amdgcn_wmma_f32_16x16x32_bf16(false, a2, false, bt, (short)0, c2, false, false);
      c3 = __builtin_amdgcn_wmma_f32_16x16x32_bf16(false, a3, false, bt, (short)0, c3, false, false);
    }
    const int rb = (lane >> 4) << 3;            // row base inside 16x16 tile
    const int c = cOff + nt * 16 + (lane & 15); // output column
#pragma unroll
    for (int i = 0; i < 8; ++i) {
      Cb[( 0 + rb + i) * cCols + c] = c0[i];
      Cb[(16 + rb + i) * cCols + c] = c1[i];
      Cb[(32 + rb + i) * cCols + c] = c2[i];
      Cb[(48 + rb + i) * cCols + c] = c3[i];
    }
  }
}

// ---------------------------------------------------------------------------
// Q = [E(e) | KC(kc)] @ Wq + bq  -> bf16 [B*S, 128]
// ---------------------------------------------------------------------------
__global__ __launch_bounds__(256)
void q_gemm(const int* __restrict__ e_data, const int* __restrict__ kc_data,
            const float* __restrict__ exE, const float* __restrict__ skE,
            const bf16* __restrict__ WqP, const float* __restrict__ bq,
            bf16* __restrict__ Qbf) {
  __shared__ __align__(32) bf16 Ab[4 * 8 * 512];
  __shared__ __align__(16) float Cb[64 * 128];
  const int tid = threadIdx.x;
  const int rowBase = blockIdx.x * 64;
  for (int p = tid; p < 64 * 128; p += 256) {           // (row, k-pair)
    int m = p >> 7, col = (p & 127) << 1;
    int g = rowBase + m;
    float x0, x1;
    if (col < 128) {
      const float* r = exE + (size_t)e_data[g] * 128 + col; x0 = r[0]; x1 = r[1];
    } else {
      const float* r = skE + (size_t)kc_data[g] * 128 + (col - 128); x0 = r[0]; x1 = r[1];
    }
    a_store2(Ab, 8, m, col, x0, x1);
  }
  __syncthreads();
  gemm64(Ab, 8, WqP, 8, 8, Cb, 128, 0);
  __syncthreads();
  for (int i = tid; i < 64 * 128; i += 256) {
    int m = i >> 7, col = i & 127;
    Qbf[(size_t)(rowBase + m) * 128 + col] = f2bf(Cb[i] + bq[col]);
  }
}

// ---------------------------------------------------------------------------
// X = [Q | Ans(a) | AT(at)] @ Wx + bx  -> bf16 [B*S, 128]
// ---------------------------------------------------------------------------
__global__ __launch_bounds__(256)
void x_gemm(const int* __restrict__ a_data, const int* __restrict__ at_data,
            const bf16* __restrict__ Qbf, const float* __restrict__ ansE,
            const float* __restrict__ atE, const bf16* __restrict__ WxP,
            const float* __restrict__ bx, bf16* __restrict__ Xbf) {
  __shared__ __align__(32) bf16 Ab[4 * 8 * 512];
  __shared__ __align__(16) float Cb[64 * 128];
  const int tid = threadIdx.x;
  const int rowBase = blockIdx.x * 64;
  for (int p = tid; p < 64 * 128; p += 256) {
    int m = p >> 7, col = (p & 127) << 1;
    int g = rowBase + m;
    if (col < 128) {
      uint32_t two = *(const uint32_t*)(Qbf + (size_t)g * 128 + col);
      a_store2bf(Ab, 8, m, col, two);
    } else if (col < 192) {
      const float* r = ansE + (size_t)a_data[g] * 64 + (col - 128);
      a_store2(Ab, 8, m, col, r[0], r[1]);
    } else {
      const float* r = atE + (size_t)at_data[g] * 64 + (col - 192);
      a_store2(Ab, 8, m, col, r[0], r[1]);
    }
  }
  __syncthreads();
  gemm64(Ab, 8, WxP, 8, 8, Cb, 128, 0);
  __syncthreads();
  for (int i = tid; i < 64 * 128; i += 256) {
    int m = i >> 7, col = i & 127;
    Xbf[(size_t)(rowBase + m) * 128 + col] = f2bf(Cb[i] + bx[col]);
  }
}

// ---------------------------------------------------------------------------
// Persistent recurrent kernel: 8 WGs x 64 batch rows, 500 serial steps.
// ---------------------------------------------------------------------------
struct RParams {
  const int*   it_data;
  const float* it_emb;
  const bf16 *Qbf, *Xbf;
  const bf16 *WuP, *Wp1P, *WzP, *WrP, *WhP, *WogP, *WovP;
  const float *bu, *bp1, *Wp2, *bp2, *bz, *br, *bh, *bog, *bov;
  const float *h0, *v0;
  float* out;
};

__global__ __launch_bounds__(256)
void okt_recur(RParams P) {
  __shared__ __align__(16) float hS [64 * 128];   // prev h (f32 state)
  __shared__ __align__(16) float vS [64 * 128];   // transfer vector v
  __shared__ __align__(16) float uhS[64 * 128];   // uh, later h_new
  __shared__ __align__(32) bf16  itS[64 * 128];
  __shared__ __align__(32) bf16  xS [64 * 128];
  __shared__ __align__(32) bf16  Ab [4 * 12 * 512];
  __shared__ __align__(16) float Cb [64 * 256];
  __shared__ float red[64 * 4];
  __shared__ int   itIdx[64];

  const int tid = threadIdx.x;
  const int rowBase = blockIdx.x * 64;

  for (int i = tid; i < 64 * 128; i += 256) {
    hS[i] = P.h0[i & 127];
    vS[i] = P.v0[i & 127];
  }
  __syncthreads();

  for (int s = 0; s < S_; ++s) {
    // ---- stage it-embeddings (bf16) and x (bf16) for this step ----
    if (tid < 64) itIdx[tid] = P.it_data[(size_t)(rowBase + tid) * S_ + s];
    __syncthreads();
    for (int i = tid; i < 64 * 128; i += 256) {
      int m = i >> 7, col = i & 127;
      itS[i] = f2bf(P.it_emb[(size_t)itIdx[m] * 128 + col]);
    }
    for (int p = tid; p < 64 * 64; p += 256) {
      int m = p >> 6, cp = (p & 63) << 1;
      *(uint32_t*)(xS + m * 128 + cp) =
        *(const uint32_t*)(P.Xbf + ((size_t)(rowBase + m) * S_ + s) * 128 + cp);
    }
    __syncthreads();

    // ---- UKSE gate: gamma = sigm([h|it] @ Wu + bu); uh = h + gamma*v ----
    for (int p = tid; p < 64 * 128; p += 256) {
      int m = p >> 7, col = (p & 127) << 1;
      if (col < 128) a_store2(Ab, 12, m, col, hS[m * 128 + col], hS[m * 128 + col + 1]);
      else a_store2bf(Ab, 12, m, col, *(const uint32_t*)(itS + m * 128 + (col - 128)));
    }
    __syncthreads();
    gemm64(Ab, 12, P.WuP, 8, 8, Cb, 256, 0);
    __syncthreads();
    for (int i = tid; i < 64 * 128; i += 256) {
      int m = i >> 7, col = i & 127;
      float g = sigm(Cb[m * 256 + col] + P.bu[col]);
      uhS[i] = hS[i] + g * vS[i];
    }
    __syncthreads();

    // ---- predict: sigm(relu([uh|q] @ Wp1 + bp1) @ Wp2 + bp2) ----
    for (int p = tid; p < 64 * 128; p += 256) {
      int m = p >> 7, col = (p & 127) << 1;
      if (col < 128) a_store2(Ab, 12, m, col, uhS[m * 128 + col], uhS[m * 128 + col + 1]);
      else a_store2bf(Ab, 12, m, col,
             *(const uint32_t*)(P.Qbf + ((size_t)(rowBase + m) * S_ + s) * 128 + (col - 128)));
    }
    __syncthreads();
    gemm64(Ab, 12, P.Wp1P, 8, 16, Cb, 256, 0);
    __syncthreads();
    {
      int row = tid >> 2, part = tid & 3;
      const float* cr = Cb + row * 256;
      float acc = 0.f;
      for (int c = part * 64; c < part * 64 + 64; ++c)
        acc += fmaxf(cr[c] + P.bp1[c], 0.f) * P.Wp2[c];
      red[row * 4 + part] = acc;
    }
    __syncthreads();
    if (tid < 64) {
      float t = red[tid * 4] + red[tid * 4 + 1] + red[tid * 4 + 2] + red[tid * 4 + 3] + P.bp2[0];
      P.out[(size_t)(rowBase + tid) * S_ + s] = sigm(t);
    }
    __syncthreads();

    // ---- KSE: z,r from [x|uh]; hc from [x|r*uh]; h_new ----
    for (int p = tid; p < 64 * 128; p += 256) {
      int m = p >> 7, col = (p & 127) << 1;
      if (col < 128) a_store2bf(Ab, 12, m, col, *(const uint32_t*)(xS + m * 128 + col));
      else a_store2(Ab, 12, m, col, uhS[m * 128 + col - 128], uhS[m * 128 + col - 127]);
    }
    __syncthreads();
    gemm64(Ab, 12, P.WzP, 8, 8, Cb, 256, 0);     // z-pre -> cols 0..127
    gemm64(Ab, 12, P.WrP, 8, 8, Cb, 256, 128);   // r-pre -> cols 128..255
    __syncthreads();
    for (int p = tid; p < 64 * 64; p += 256) {   // overwrite A k=128.. with r*uh
      int m = p >> 6, col = (p & 63) << 1;
      float r0 = sigm(Cb[m * 256 + 128 + col]     + P.br[col]);
      float r1 = sigm(Cb[m * 256 + 128 + col + 1] + P.br[col + 1]);
      a_store2(Ab, 12, m, 128 + col, r0 * uhS[m * 128 + col], r1 * uhS[m * 128 + col + 1]);
    }
    __syncthreads();
    gemm64(Ab, 12, P.WhP, 8, 8, Cb, 256, 128);   // hc-pre -> cols 128..255 (z kept)
    __syncthreads();
    for (int i = tid; i < 64 * 128; i += 256) {  // h_new into uhS
      int m = i >> 7, col = i & 127;
      float z  = sigm (Cb[m * 256 + col]       + P.bz[col]);
      float hc = tanhf(Cb[m * 256 + 128 + col] + P.bh[col]);
      uhS[i] = (1.f - z) * uhS[i] + z * hc;
    }
    __syncthreads();

    // ---- OTE: o_in = [prev_h | h_new | it]; g, v_new ----
    for (int p = tid; p < 64 * 192; p += 256) {
      int m = p / 192, col = (p - m * 192) << 1;
      if (col < 128)      a_store2(Ab, 12, m, col, hS[m * 128 + col], hS[m * 128 + col + 1]);
      else if (col < 256) a_store2(Ab, 12, m, col, uhS[m * 128 + col - 128], uhS[m * 128 + col - 127]);
      else a_store2bf(Ab, 12, m, col, *(const uint32_t*)(itS + m * 128 + (col - 256)));
    }
    __syncthreads();
    gemm64(Ab, 12, P.WogP, 12, 8, Cb, 256, 0);
    gemm64(Ab, 12, P.WovP, 12, 8, Cb, 256, 128);
    __syncthreads();
    for (int i = tid; i < 64 * 128; i += 256) {
      int m = i >> 7, col = i & 127;
      float g  = sigm (Cb[m * 256 + col]       + P.bog[col]);
      float ov = tanhf(Cb[m * 256 + 128 + col] + P.bov[col]);
      vS[i] = g * vS[i] + (1.f - g) * ov;
      hS[i] = uhS[i];                          // commit h_new
    }
    __syncthreads();
  }
}

// ---------------------------------------------------------------------------
extern "C" void kernel_launch(void* const* d_in, const int* in_sizes, int n_in,
                              void* d_out, int out_size, void* d_ws, size_t ws_size,
                              hipStream_t stream) {
  (void)in_sizes; (void)n_in; (void)out_size; (void)ws_size;
  const int*   kc   = (const int*)d_in[0];
  const int*   a    = (const int*)d_in[1];
  const int*   e    = (const int*)d_in[2];
  const int*   it   = (const int*)d_in[3];
  const int*   at   = (const int*)d_in[4];
  const float* itE  = (const float*)d_in[5];
  const float* atE  = (const float*)d_in[6];
  const float* ansE = (const float*)d_in[7];
  const float* exE  = (const float*)d_in[8];
  const float* skE  = (const float*)d_in[9];
  const float* Wq = (const float*)d_in[10]; const float* bq = (const float*)d_in[11];
  const float* Wx = (const float*)d_in[12]; const float* bx = (const float*)d_in[13];
  const float* Wu = (const float*)d_in[14]; const float* bu = (const float*)d_in[15];
  const float* Wz = (const float*)d_in[16]; const float* bz = (const float*)d_in[17];
  const float* Wr = (const float*)d_in[18]; const float* br = (const float*)d_in[19];
  const float* Wh = (const float*)d_in[20]; const float* bh = (const float*)d_in[21];
  const float* Wog= (const float*)d_in[22]; const float* bog= (const float*)d_in[23];
  const float* Wov= (const float*)d_in[24]; const float* bov= (const float*)d_in[25];
  const float* Wp1= (const float*)d_in[26]; const float* bp1= (const float*)d_in[27];
  const float* Wp2= (const float*)d_in[28]; const float* bp2= (const float*)d_in[29];
  const float* h0 = (const float*)d_in[30]; const float* v0 = (const float*)d_in[31];

  // workspace layout (bf16 elements)
  bf16* ws = (bf16*)d_ws;
  bf16 *WqP  = ws + 0;
  bf16 *WxP  = ws + 32768;
  bf16 *WuP  = ws + 65536;
  bf16 *WzP  = ws + 98304;
  bf16 *WrP  = ws + 131072;
  bf16 *WhP  = ws + 163840;
  bf16 *WogP = ws + 196608;
  bf16 *WovP = ws + 245760;
  bf16 *Wp1P = ws + 294912;
  bf16 *Qbf  = ws + 360448;                       // [B*S,128] bf16
  bf16 *Xbf  = Qbf + (size_t)B_ * S_ * 128;       // [B*S,128] bf16

  pack_b<<<128, 256, 0, stream>>>(Wq,  WqP,  256, 128);
  pack_b<<<128, 256, 0, stream>>>(Wx,  WxP,  256, 128);
  pack_b<<<128, 256, 0, stream>>>(Wu,  WuP,  256, 128);
  pack_b<<<128, 256, 0, stream>>>(Wz,  WzP,  256, 128);
  pack_b<<<128, 256, 0, stream>>>(Wr,  WrP,  256, 128);
  pack_b<<<128, 256, 0, stream>>>(Wh,  WhP,  256, 128);
  pack_b<<<192, 256, 0, stream>>>(Wog, WogP, 384, 128);
  pack_b<<<192, 256, 0, stream>>>(Wov, WovP, 384, 128);
  pack_b<<<256, 256, 0, stream>>>(Wp1, Wp1P, 256, 256);

  q_gemm<<<(B_ * S_) / 64, 256, 0, stream>>>(e, kc, exE, skE, WqP, bq, Qbf);
  x_gemm<<<(B_ * S_) / 64, 256, 0, stream>>>(a, at, Qbf, ansE, atE, WxP, bx, Xbf);

  RParams P{it, itE, Qbf, Xbf,
            WuP, Wp1P, WzP, WrP, WhP, WogP, WovP,
            bu, bp1, Wp2, bp2, bz, br, bh, bog, bov,
            h0, v0, (float*)d_out};
  okt_recur<<<B_ / 64, 256, 0, stream>>>(P);
}
